// InCTRL_88742614270286
// MI455X (gfx1250) — matrix operand order repaired and checked by reading, not compiled, not run on hardware
//
#include <hip/hip_runtime.h>
#include <math.h>

typedef __bf16 v8bf  __attribute__((ext_vector_type(8)));
typedef __bf16 v16bf __attribute__((ext_vector_type(16)));
typedef float  v8f   __attribute__((ext_vector_type(8)));

#define BB   64
#define PP   225
#define KK   16
#define DD   640
#define MROWS (BB*PP)      // 14400
#define NROWS (KK*PP)      // 3600
#define NTILE 128          // N tile per block
#define NTILES 29          // ceil(3600/128)
#define KTILES (DD/64)     // 10 K-tiles of 64
#define LDT  72            // padded LDS tile stride (halfwords); 144B rows, 16B aligned
#define A_OFF (64*LDT)     // B tile offset inside one LDS buffer (halfwords)
#define TILE_HW ((64+128)*LDT)

// ---- CDNA5 async global->LDS path (guarded; falls back to sync copy) ------
#if defined(__has_builtin)
#  if __has_builtin(__builtin_amdgcn_global_load_async_to_lds_b128)
#    define HAVE_ASYNC_LDS 1
#  endif
#endif
#ifndef HAVE_ASYNC_LDS
#  define HAVE_ASYNC_LDS 0
#endif

// builtin signature (from compiler diagnostic):
//   (v4i __global*, v4i __shared*, Ii offset, Ii cpol)  with v4i = gcc-vector int[4]
typedef int v4i_vs __attribute__((__vector_size__(4 * sizeof(int))));
typedef __attribute__((address_space(1))) v4i_vs as1_v4i;
typedef __attribute__((address_space(3))) v4i_vs as3_v4i;

__device__ __forceinline__ void async_wait_all() {
#if HAVE_ASYNC_LDS
#  if __has_builtin(__builtin_amdgcn_s_wait_asynccnt)
  __builtin_amdgcn_s_wait_asynccnt(0);
#  else
  asm volatile("s_wait_asynccnt 0x0" ::: "memory");
#  endif
#endif
}

__device__ __forceinline__ void cp16_g2l(__bf16* l, const __bf16* g) {
#if HAVE_ASYNC_LDS
  __builtin_amdgcn_global_load_async_to_lds_b128(
      (as1_v4i*)(v4i_vs*)(void*)g,
      (as3_v4i*)(v4i_vs*)(void*)l, 0, 0);
#else
  *(uint4*)l = *(const uint4*)g;
#endif
}

__device__ __forceinline__ v16bf cat16(v8bf lo, v8bf hi) {
  return __builtin_shufflevector(lo, hi, 0,1,2,3,4,5,6,7,8,9,10,11,12,13,14,15);
}

// ---------------------------------------------------------------------------
// 1) L2-normalize rows of q_patch (14400 x 640) and r_patch (3600 x 640),
//    emit bf16 for WMMA.  One block per row, 128 threads x 5 elems.
// ---------------------------------------------------------------------------
__global__ __launch_bounds__(128) void nrm_kernel(const float* __restrict__ q,
                                                  const float* __restrict__ r,
                                                  unsigned short* __restrict__ qn_u,
                                                  unsigned short* __restrict__ rn_u) {
  __bf16* qn = (__bf16*)qn_u;
  __bf16* rn = (__bf16*)rn_u;
  const int row = blockIdx.x;
  const float* src;
  __bf16* dst;
  if (row < MROWS) { src = q + (size_t)row * DD;           dst = qn + (size_t)row * DD; }
  else             { src = r + (size_t)(row - MROWS) * DD; dst = rn + (size_t)(row - MROWS) * DD; }

  const int t = threadIdx.x;
  float x[5]; float ss = 0.f;
#pragma unroll
  for (int i = 0; i < 5; ++i) { x[i] = src[t + i * 128]; ss += x[i] * x[i]; }
#pragma unroll
  for (int m = 1; m < 32; m <<= 1) ss += __shfl_xor(ss, m);
  __shared__ float ps[4];
  if ((t & 31) == 0) ps[t >> 5] = ss;
  __syncthreads();
  float tot = ps[0] + ps[1] + ps[2] + ps[3];
  float scale = 1.0f / (sqrtf(tot) + 1e-6f);
#pragma unroll
  for (int i = 0; i < 5; ++i) dst[t + i * 128] = (__bf16)(x[i] * scale);
}

// ---------------------------------------------------------------------------
// 2) Adapter MLP: fa_mean = mean_k relu(relu(r_img @ w1) @ w2)   [640]
// ---------------------------------------------------------------------------
__global__ __launch_bounds__(256) void adpt_kernel(const float* __restrict__ rimg,
                                                   const float* __restrict__ w1,
                                                   const float* __restrict__ w2,
                                                   float* __restrict__ famean) {
  __shared__ float h[KK * 160];
  const int t = threadIdx.x;
  for (int idx = t; idx < KK * 160; idx += 256) {
    int k = idx / 160, j = idx - k * 160;
    float a = 0.f;
    for (int d = 0; d < DD; ++d) a += rimg[k * DD + d] * w1[d * 160 + j];
    h[idx] = fmaxf(a, 0.f);
  }
  __syncthreads();
  for (int d = t; d < DD; d += 256) {
    float acc = 0.f;
    for (int k = 0; k < KK; ++k) {
      float s = 0.f;
      for (int j = 0; j < 160; ++j) s += h[k * 160 + j] * w2[j * DD + d];
      acc += fmaxf(s, 0.f);
    }
    famean[d] = acc * (1.0f / KK);
  }
}

// ---------------------------------------------------------------------------
// 3) Fused bf16 WMMA GEMM (qn @ rn^T) + per-row max -> amap = 0.5*(1-max)
//    Block: M=64 x N=128 tile, 8 waves as 2(wm) x 4(wn), 32x32 per wave
//    (4 accumulators).  K tiles of 64 double-buffered through LDS with
//    async global->LDS copies overlapping WMMA.
// ---------------------------------------------------------------------------
__device__ __forceinline__ v16bf ldA(const __bf16* As, int row, int koff, int k0) {
  const __bf16* p = As + row * LDT + koff + k0;       // A frag: chunks at +0,+16
  v8bf lo = *(const v8bf*)p;
  v8bf hi = *(const v8bf*)(p + 16);
  return cat16(lo, hi);
}
__device__ __forceinline__ v16bf ldB(const __bf16* Bs, int row, int koff, int k0) {
  const __bf16* p = Bs + row * LDT + koff + k0;       // B frag: chunks at +0,+16
  v8bf lo = *(const v8bf*)p;
  v8bf hi = *(const v8bf*)(p + 16);
  return cat16(lo, hi);
}

// cooperative async issue of one K-tile: A 64x64 (512 x 16B) + B 128x64 (1024 x 16B)
__device__ __forceinline__ void issue_tile(const __bf16* __restrict__ qn,
                                           const __bf16* __restrict__ rn,
                                           __bf16* __restrict__ buf,
                                           int m0, int n0, int kbase, int t) {
#pragma unroll
  for (int c = 0; c < 6; ++c) {
    int idx = t + c * 256;                 // 0..1535, uniform branch per c
    if (idx < 512) {                       // A tile
      int row = idx >> 3, ch = idx & 7;
      cp16_g2l(buf + row * LDT + ch * 8,
               qn + (size_t)(m0 + row) * DD + kbase + ch * 8);
    } else {                               // B tile
      int i2 = idx - 512;
      int row = i2 >> 3, ch = i2 & 7;
      int gr = n0 + row;
      if (gr < NROWS)
        cp16_g2l(buf + A_OFF + row * LDT + ch * 8,
                 rn + (size_t)gr * DD + kbase + ch * 8);
    }
  }
}

__global__ __launch_bounds__(256) void simmax_kernel(const unsigned short* __restrict__ qn_u,
                                                     const unsigned short* __restrict__ rn_u,
                                                     float* __restrict__ amap) {
  const __bf16* qn = (const __bf16*)qn_u;
  const __bf16* rn = (const __bf16*)rn_u;
  __shared__ __bf16 tiles[2][TILE_HW];     // 2 x (64+128) x 72 bf16 = 55.3 KB
  __shared__ float rowpart[4][64];

  const int t    = threadIdx.x;
  const int lane = t & 31;
  const int wave = t >> 5;
  const int wm   = wave & 1;    // 32-row M sub-tile
  const int wn   = wave >> 1;   // 32-col N group (0..3)
  const int lh   = lane & 15;
  const int hi16 = (lane >= 16) ? 8 : 0;
  const int m0   = blockIdx.x * 64;

  // ISA fragment K-offsets (halfwords within a 32-K step)
  const int a_k0 = (lane < 16) ? 0 : 8;   // A: chunks at k0 and k0+16
  const int b_k0 = (lane < 16) ? 0 : 8;   // B: chunks at k0 and k0+16

  float rm[2][8];
#pragma unroll
  for (int i = 0; i < 2; ++i)
#pragma unroll
    for (int r = 0; r < 8; ++r) rm[i][r] = -1e30f;

  for (int nt = 0; nt < NTILES; ++nt) {
    const int n0 = nt * NTILE;
    v8f c00 = {}, c01 = {}, c10 = {}, c11 = {};

    issue_tile(qn, rn, &tiles[0][0], m0, n0, 0, t);      // prologue: K-tile 0

    for (int kt = 0; kt < KTILES; ++kt) {
      async_wait_all();          // my async loads for tile kt have landed
      __syncthreads();           // everyone's have landed
      if (kt + 1 < KTILES)       // overlap next tile's loads with compute
        issue_tile(qn, rn, &tiles[(kt + 1) & 1][0], m0, n0, (kt + 1) * 64, t);

      const __bf16* As = &tiles[kt & 1][0];
      const __bf16* Bs = &tiles[kt & 1][A_OFF];
#pragma unroll
      for (int ks = 0; ks < 2; ++ks) {
        const int koff = ks * 32;
        v16bf a0 = ldA(As, wm * 32 + lh,      koff, a_k0);
        v16bf a1 = ldA(As, wm * 32 + 16 + lh, koff, a_k0);
        v16bf b0 = ldB(Bs, wn * 32 + lh,      koff, b_k0);
        v16bf b1 = ldB(Bs, wn * 32 + 16 + lh, koff, b_k0);
        c00 = __builtin_amdgcn_wmma_f32_16x16x32_bf16(false, a0, false, b0, (short)0, c00, false, false);
        c01 = __builtin_amdgcn_wmma_f32_16x16x32_bf16(false, a0, false, b1, (short)0, c01, false, false);
        c10 = __builtin_amdgcn_wmma_f32_16x16x32_bf16(false, a1, false, b0, (short)0, c10, false, false);
        c11 = __builtin_amdgcn_wmma_f32_16x16x32_bf16(false, a1, false, b1, (short)0, c11, false, false);
      }
    }

    // fold this N tile into the running per-row max.
    // C layout: lane&15 = N, vgpr r = M (lanes>=16 -> M+8)
    const bool ok0 = (n0 + wn * 32 + lh)      < NROWS;
    const bool ok1 = (n0 + wn * 32 + 16 + lh) < NROWS;
#pragma unroll
    for (int r = 0; r < 8; ++r) {
      {
        float v = fmaxf(ok0 ? c00[r] : -1e30f, ok1 ? c01[r] : -1e30f);
        v = fmaxf(v, __shfl_xor(v, 1));
        v = fmaxf(v, __shfl_xor(v, 2));
        v = fmaxf(v, __shfl_xor(v, 4));
        v = fmaxf(v, __shfl_xor(v, 8));
        rm[0][r] = fmaxf(rm[0][r], v);
      }
      {
        float v = fmaxf(ok0 ? c10[r] : -1e30f, ok1 ? c11[r] : -1e30f);
        v = fmaxf(v, __shfl_xor(v, 1));
        v = fmaxf(v, __shfl_xor(v, 2));
        v = fmaxf(v, __shfl_xor(v, 4));
        v = fmaxf(v, __shfl_xor(v, 8));
        rm[1][r] = fmaxf(rm[1][r], v);
      }
    }
  }

  __syncthreads();
  if (lh == 0) {
#pragma unroll
    for (int i = 0; i < 2; ++i)
#pragma unroll
      for (int r = 0; r < 8; ++r)
        rowpart[wn][wm * 32 + i * 16 + r + hi16] = rm[i][r];
  }
  __syncthreads();
  if (t < 64) {
    float m = fmaxf(fmaxf(rowpart[0][t], rowpart[1][t]),
                    fmaxf(rowpart[2][t], rowpart[3][t]));
    amap[m0 + t] = 0.5f * (1.0f - m);
  }
}

// ---------------------------------------------------------------------------
// 4) Both 3-layer heads + mean(amap) + final score.  One block per sample.
// ---------------------------------------------------------------------------
__global__ __launch_bounds__(128) void heads_kernel(
    const float* __restrict__ qimg, const float* __restrict__ famean,
    const float* __restrict__ amap,
    const float* dh_w1, const float* dh_b1, const float* dh_g2, const float* dh_be2,
    const float* dh_w2, const float* dh_b2, const float* dh_g3, const float* dh_be3,
    const float* dh_w3, const float* dh_b3,
    const float* dr_w1, const float* dr_b1, const float* dr_g2, const float* dr_be2,
    const float* dr_w2, const float* dr_b2, const float* dr_g3, const float* dr_be3,
    const float* dr_w3, const float* dr_b3,
    float* __restrict__ out) {
  __shared__ float res[DD];
  __shared__ float am[PP];
  __shared__ float h1[128];
  __shared__ float h2[64];
  __shared__ float red[128];
  __shared__ float sref_s, smap_s;

  const int b = blockIdx.x, t = threadIdx.x;
  for (int d = t; d < DD; d += 128) res[d] = qimg[b * DD + d] - famean[d];
  for (int p = t; p < PP; p += 128) am[p] = amap[b * PP + p];
  __syncthreads();

  // ---- ref head: 640 -> 128 -> 64 -> 1 ----
  {
    float a = dr_b1[t];
    for (int d = 0; d < DD; ++d) a += res[d] * dr_w1[d * 128 + t];
    h1[t] = fmaxf(a, 0.f) * dr_g2[t] + dr_be2[t];
  }
  __syncthreads();
  if (t < 64) {
    float a = dr_b2[t];
    for (int i = 0; i < 128; ++i) a += h1[i] * dr_w2[i * 64 + t];
    h2[t] = fmaxf(a, 0.f) * dr_g3[t] + dr_be3[t];
  }
  __syncthreads();
  red[t] = (t < 64) ? h2[t] * dr_w3[t] : 0.f;
  __syncthreads();
  for (int s = 64; s > 0; s >>= 1) { if (t < s) red[t] += red[t + s]; __syncthreads(); }
  if (t == 0) sref_s = 1.0f / (1.0f + expf(-(red[0] + dr_b3[0])));
  __syncthreads();

  // ---- map head: 225 -> 128 -> 64 -> 1 ----
  {
    float a = dh_b1[t];
    for (int p = 0; p < PP; ++p) a += am[p] * dh_w1[p * 128 + t];
    h1[t] = fmaxf(a, 0.f) * dh_g2[t] + dh_be2[t];
  }
  __syncthreads();
  if (t < 64) {
    float a = dh_b2[t];
    for (int i = 0; i < 128; ++i) a += h1[i] * dh_w2[i * 64 + t];
    h2[t] = fmaxf(a, 0.f) * dh_g3[t] + dh_be3[t];
  }
  __syncthreads();
  red[t] = (t < 64) ? h2[t] * dh_w3[t] : 0.f;
  __syncthreads();
  for (int s = 64; s > 0; s >>= 1) { if (t < s) red[t] += red[t + s]; __syncthreads(); }
  if (t == 0) smap_s = 1.0f / (1.0f + expf(-(red[0] + dh_b3[0])));
  __syncthreads();

  // ---- mean(amap) + combine ----
  float pm = 0.f;
  for (int p = t; p < PP; p += 128) pm += am[p];
  red[t] = pm;
  __syncthreads();
  for (int s = 64; s > 0; s >>= 1) { if (t < s) red[t] += red[t + s]; __syncthreads(); }
  if (t == 0) out[b] = 0.5f * (sref_s + smap_s) + red[0] * (1.0f / PP);
}

// ---------------------------------------------------------------------------
extern "C" void kernel_launch(void* const* d_in, const int* in_sizes, int n_in,
                              void* d_out, int out_size, void* d_ws, size_t ws_size,
                              hipStream_t stream) {
  const float* q_patch = (const float*)d_in[0];
  const float* r_patch = (const float*)d_in[1];
  const float* q_img   = (const float*)d_in[2];
  const float* r_img   = (const float*)d_in[3];
  const float* adpt_w1 = (const float*)d_in[4];
  const float* adpt_w2 = (const float*)d_in[5];
  const float* dh_w1 = (const float*)d_in[6];   const float* dh_b1 = (const float*)d_in[7];
  const float* dh_g2 = (const float*)d_in[8];   const float* dh_be2 = (const float*)d_in[9];
  const float* dh_w2 = (const float*)d_in[10];  const float* dh_b2 = (const float*)d_in[11];
  const float* dh_g3 = (const float*)d_in[12];  const float* dh_be3 = (const float*)d_in[13];
  const float* dh_w3 = (const float*)d_in[14];  const float* dh_b3 = (const float*)d_in[15];
  const float* dr_w1 = (const float*)d_in[16];  const float* dr_b1 = (const float*)d_in[17];
  const float* dr_g2 = (const float*)d_in[18];  const float* dr_be2 = (const float*)d_in[19];
  const float* dr_w2 = (const float*)d_in[20];  const float* dr_b2 = (const float*)d_in[21];
  const float* dr_g3 = (const float*)d_in[22];  const float* dr_be3 = (const float*)d_in[23];
  const float* dr_w3 = (const float*)d_in[24];  const float* dr_b3 = (const float*)d_in[25];

  // workspace layout (all 16B aligned)
  char* ws = (char*)d_ws;
  unsigned short* qn = (unsigned short*)ws;                         // 14400*640*2 = 18,432,000
  unsigned short* rn = (unsigned short*)(ws + 18432000);            //  3600*640*2 =  4,608,000
  float* amap   = (float*)(ws + 18432000 + 4608000);                // 14400*4     =     57,600
  float* famean = (float*)(ws + 18432000 + 4608000 + 57600);        //   640*4     =      2,560

  nrm_kernel<<<MROWS + NROWS, 128, 0, stream>>>(q_patch, r_patch, qn, rn);
  adpt_kernel<<<1, 256, 0, stream>>>(r_img, adpt_w1, adpt_w2, famean);
  simmax_kernel<<<MROWS / 64, 256, 0, stream>>>(qn, rn, amap);
  heads_kernel<<<BB, 128, 0, stream>>>(q_img, famean, amap,
                                       dh_w1, dh_b1, dh_g2, dh_be2, dh_w2, dh_b2, dh_g3, dh_be3, dh_w3, dh_b3,
                                       dr_w1, dr_b1, dr_g2, dr_be2, dr_w2, dr_b2, dr_g3, dr_be3, dr_w3, dr_b3,
                                       (float*)d_out);
}
